// EdgeAggr_15470472201061
// MI455X (gfx1250) — compile-verified, matrix operand order checked by reading, not run on hardware
//
#include <hip/hip_runtime.h>
#include <hip/hip_bf16.h>
#include <stdint.h>

// ---------------- CDNA5 (gfx1250) types ----------------
typedef __attribute__((ext_vector_type(16))) __bf16 v16bf;
typedef __attribute__((ext_vector_type(8)))  float  v8f;

#define D 128
#define HARD_C 10000
#define NBITS  (HARD_C * HARD_C)          // 100,000,000 possible keys
#define NWORDS ((NBITS + 31) / 32)        // 3,125,000 bitmap words
#define SCAN_TPB 256
#define WPT 8
#define WORDS_PER_BLOCK (SCAN_TPB * WPT)  // 2048
#define NBLK ((NWORDS + WORDS_PER_BLOCK - 1) / WORDS_PER_BLOCK)  // 1526
#define WST 136                            // LDS row stride (bf16) -> 16B aligned rows

static __device__ __forceinline__ float gelu_tanh(float v) {
    return 0.5f * v * (1.0f + tanhf(0.7978845608028654f * (v + 0.044715f * v * v * v)));
}

// ---------------- utility: zero a u32 region (ws is poisoned 0xAA) ----------------
__global__ void zero_u32(unsigned* __restrict__ p, size_t n) {
    size_t i = (size_t)blockIdx.x * blockDim.x + threadIdx.x;
    size_t stride = (size_t)gridDim.x * blockDim.x;
    for (; i < n; i += stride) p[i] = 0u;
}

// ---------------- 1) mark unique keys in a 100M-bit bitmap ----------------
__global__ void set_bits(const int* __restrict__ row, const int* __restrict__ col,
                         const int* __restrict__ cptr, unsigned* __restrict__ bitmap, int E) {
    int e = blockIdx.x * blockDim.x + threadIdx.x;
    if (e >= E) return;
    int c = cptr[0];
    unsigned key = (unsigned)row[e] * (unsigned)c + (unsigned)col[e];
    atomicOr(&bitmap[key >> 5], 1u << (key & 31));
}

// ---------------- 2) popcount scan: per-block partial sums ----------------
__global__ void scan_partial(const unsigned* __restrict__ bitmap,
                             unsigned* __restrict__ bsums, int nwords) {
    __shared__ unsigned sd[SCAN_TPB];
    int t = threadIdx.x;
    int base = blockIdx.x * WORDS_PER_BLOCK + t * WPT;
    unsigned s = 0;
#pragma unroll
    for (int j = 0; j < WPT; j++) {
        int w = base + j;
        if (w < nwords) s += __popc(bitmap[w]);
    }
    sd[t] = s;
    __syncthreads();
    for (int off = SCAN_TPB / 2; off > 0; off >>= 1) {
        if (t < off) sd[t] += sd[t + off];
        __syncthreads();
    }
    if (t == 0) bsums[blockIdx.x] = sd[0];
}

// ---------------- 3) serial exclusive scan of 1526 block sums ----------------
__global__ void scan_bsums(unsigned* __restrict__ bsums, int nblk) {
    if (blockIdx.x == 0 && threadIdx.x == 0) {
        unsigned run = 0;
        for (int i = 0; i < nblk; i++) { unsigned v = bsums[i]; bsums[i] = run; run += v; }
        bsums[nblk] = run;  // total unique count
    }
}

// ---------------- 4) word-level exclusive prefix of set-bit counts ----------------
__global__ void scan_words(const unsigned* __restrict__ bitmap, const unsigned* __restrict__ bsums,
                           unsigned* __restrict__ prefix, int nwords) {
    __shared__ unsigned sd[SCAN_TPB];
    int t = threadIdx.x;
    int base = blockIdx.x * WORDS_PER_BLOCK + t * WPT;
    unsigned pc[WPT];
    unsigned s = 0;
#pragma unroll
    for (int j = 0; j < WPT; j++) {
        int w = base + j;
        pc[j] = (w < nwords) ? __popc(bitmap[w]) : 0u;
        s += pc[j];
    }
    sd[t] = s;
    __syncthreads();
    // Hillis-Steele inclusive scan of thread totals
    for (int off = 1; off < SCAN_TPB; off <<= 1) {
        unsigned v = (t >= off) ? sd[t - off] : 0u;
        __syncthreads();
        sd[t] += v;
        __syncthreads();
    }
    unsigned run = bsums[blockIdx.x] + (sd[t] - s);
#pragma unroll
    for (int j = 0; j < WPT; j++) {
        int w = base + j;
        if (w < nwords) prefix[w] = run;
        run += pc[j];
    }
}

// ---------------- 5) emit sorted edge_index (erow, ecol) as f32 ----------------
__global__ void emit_edges(const unsigned* __restrict__ bitmap, const unsigned* __restrict__ prefix,
                           const int* __restrict__ cptr, float* __restrict__ out, int U, int nwords) {
    int w = blockIdx.x * blockDim.x + threadIdx.x;
    if (w >= nwords) return;
    unsigned bits = bitmap[w];
    if (!bits) return;
    unsigned c = (unsigned)cptr[0];
    unsigned r = prefix[w];
    unsigned keyBase = (unsigned)w * 32u;
    while (bits) {
        int b = __ffs((int)bits) - 1;
        bits &= bits - 1u;
        unsigned key = keyBase + (unsigned)b;
        if ((int)r < U) {
            out[r] = (float)(key / c);
            out[(size_t)U + r] = (float)(key % c);
        }
        r++;
    }
}

// ---------------- 6) scatter-add x rows into agg[rank] (1 wave per edge) ----------------
__global__ void scatter_add(const float* __restrict__ x, const int* __restrict__ row,
                            const int* __restrict__ col, const int* __restrict__ cptr,
                            const unsigned* __restrict__ bitmap, const unsigned* __restrict__ prefix,
                            float* __restrict__ agg, int E) {
    int t = threadIdx.x;
    int lane = t & 31, wv = t >> 5;
    int e = blockIdx.x * 8 + wv;
    if (e >= E) return;
    int c = cptr[0];
    unsigned key = (unsigned)row[e] * (unsigned)c + (unsigned)col[e];
    unsigned w = key >> 5, b = key & 31;
    unsigned r = prefix[w] + __popc(bitmap[w] & ((1u << b) - 1u));
    const float4 xv = *(const float4*)(x + (size_t)e * D + lane * 4);
    float* ap = agg + (size_t)r * D + lane * 4;
    atomicAdd(ap + 0, xv.x);
    atomicAdd(ap + 1, xv.y);
    atomicAdd(ap + 2, xv.z);
    atomicAdd(ap + 3, xv.w);
}

// ---------------- 7) layer 1: h = gelu((pool + agg) @ W1 + b1), bf16 WMMA ----------------
__global__ __launch_bounds__(256)
void layer1_kernel(const float* __restrict__ agg, const float* __restrict__ emb,
                   const float* __restrict__ W1, const float* __restrict__ b1,
                   __bf16* __restrict__ hbuf, int U) {
    __shared__ __bf16 Wt[D * WST];   // W1 transposed [n][k], bf16
    __shared__ float embs[D];
    __shared__ float bias[D];
    int t = threadIdx.x;
    for (int idx = t; idx < D * D; idx += 256) {
        int k = idx >> 7, n = idx & 127;          // W1 row-major [k][n]
        Wt[n * WST + k] = (__bf16)W1[idx];
    }
    if (t < D) { embs[t] = emb[t]; bias[t] = b1[t]; }
    __syncthreads();

    int lane = t & 31, wv = t >> 5;
    int r0 = blockIdx.x * 128 + wv * 16;          // 16 rows per wave, 8 waves per block
    if (r0 >= U) return;                          // uniform per wave: EXEC stays all-ones
    int mlan = lane & 15, hi = lane >> 4;
    int arow = r0 + mlan; if (arow > U - 1) arow = U - 1;

    v8f acc[8];
    v8f zero = {0.f, 0.f, 0.f, 0.f, 0.f, 0.f, 0.f, 0.f};
#pragma unroll
    for (int i = 0; i < 8; i++) acc[i] = zero;

    for (int ks = 0; ks < 4; ks++) {
        int kb = ks * 32 + hi * 8;                // A layout: lanes 16-31 take K+8 halves
        const float* ap = agg + (size_t)arow * D + kb;
        v16bf a;
#pragma unroll
        for (int j = 0; j < 8; j++) {
            a[j]     = (__bf16)(ap[j]      + embs[kb + j]);
            a[8 + j] = (__bf16)(ap[16 + j] + embs[kb + 16 + j]);
        }
#pragma unroll
        for (int t8 = 0; t8 < 8; t8++) {
            const __bf16* bp = &Wt[(t8 * 16 + mlan) * WST + kb];
            v16bf b;
#pragma unroll
            for (int j = 0; j < 8; j++) { b[j] = bp[j]; b[8 + j] = bp[16 + j]; }
            acc[t8] = __builtin_amdgcn_wmma_f32_16x16x32_bf16(
                false, a, false, b, (short)0, acc[t8], false, false);
        }
    }
#pragma unroll
    for (int t8 = 0; t8 < 8; t8++) {
        int n = t8 * 16 + mlan;
#pragma unroll
        for (int i = 0; i < 8; i++) {             // C/D layout: M = i + 8*hi, N = lane&15
            int r = r0 + hi * 8 + i;
            if (r < U) {
                float v = acc[t8][i] + bias[n];
                hbuf[(size_t)r * D + n] = (__bf16)gelu_tanh(v);
            }
        }
    }
}

// ---------------- 8) layer 2: y = h @ W2 + b2, bf16 WMMA, f32 out ----------------
__global__ __launch_bounds__(256)
void layer2_kernel(const __bf16* __restrict__ hbuf, const float* __restrict__ W2,
                   const float* __restrict__ b2, float* __restrict__ y, int U) {
    __shared__ __bf16 Wt[D * WST];
    __shared__ float bias[D];
    int t = threadIdx.x;
    for (int idx = t; idx < D * D; idx += 256) {
        int k = idx >> 7, n = idx & 127;
        Wt[n * WST + k] = (__bf16)W2[idx];
    }
    if (t < D) bias[t] = b2[t];
    __syncthreads();

    int lane = t & 31, wv = t >> 5;
    int r0 = blockIdx.x * 128 + wv * 16;
    if (r0 >= U) return;
    int mlan = lane & 15, hi = lane >> 4;
    int arow = r0 + mlan; if (arow > U - 1) arow = U - 1;

    v8f acc[8];
    v8f zero = {0.f, 0.f, 0.f, 0.f, 0.f, 0.f, 0.f, 0.f};
#pragma unroll
    for (int i = 0; i < 8; i++) acc[i] = zero;

    for (int ks = 0; ks < 4; ks++) {
        int kb = ks * 32 + hi * 8;
        const __bf16* ap = hbuf + (size_t)arow * D + kb;
        v16bf a;
#pragma unroll
        for (int j = 0; j < 8; j++) { a[j] = ap[j]; a[8 + j] = ap[16 + j]; }
#pragma unroll
        for (int t8 = 0; t8 < 8; t8++) {
            const __bf16* bp = &Wt[(t8 * 16 + mlan) * WST + kb];
            v16bf b;
#pragma unroll
            for (int j = 0; j < 8; j++) { b[j] = bp[j]; b[8 + j] = bp[16 + j]; }
            acc[t8] = __builtin_amdgcn_wmma_f32_16x16x32_bf16(
                false, a, false, b, (short)0, acc[t8], false, false);
        }
    }
#pragma unroll
    for (int t8 = 0; t8 < 8; t8++) {
        int n = t8 * 16 + mlan;
#pragma unroll
        for (int i = 0; i < 8; i++) {
            int r = r0 + hi * 8 + i;
            if (r < U) y[(size_t)r * D + n] = acc[t8][i] + bias[n];
        }
    }
}

// ---------------- host launch ----------------
extern "C" void kernel_launch(void* const* d_in, const int* in_sizes, int n_in,
                              void* d_out, int out_size, void* d_ws, size_t ws_size,
                              hipStream_t stream) {
    const float* x   = (const float*)d_in[0];
    const int*   row = (const int*)d_in[1];
    const int*   col = (const int*)d_in[2];
    const int*   cp  = (const int*)d_in[3];
    const float* emb = (const float*)d_in[4];
    const float* W1  = (const float*)d_in[5];
    const float* b1  = (const float*)d_in[6];
    const float* W2  = (const float*)d_in[7];
    const float* b2  = (const float*)d_in[8];
    float* out = (float*)d_out;
    (void)n_in; (void)ws_size;

    int E = in_sizes[1];
    int U = out_size / (D + 2);     // out = [erow(U), ecol(U), y3(U*D)]

    // workspace layout: bitmap | word-prefix | block-sums | h (bf16)
    uint8_t* ws = (uint8_t*)d_ws;
    size_t off = 0;
    unsigned* bitmap = (unsigned*)(ws + off); off += (((size_t)NWORDS * 4) + 255) & ~(size_t)255;
    unsigned* prefix = (unsigned*)(ws + off); off += (((size_t)NWORDS * 4) + 255) & ~(size_t)255;
    unsigned* bsums  = (unsigned*)(ws + off); off += (((size_t)(NBLK + 1) * 4) + 255) & ~(size_t)255;
    __bf16*   hbuf   = (__bf16*)(ws + off);

    float* agg = out + (size_t)2 * U;   // accumulate into y3 region, overwritten by layer2

    zero_u32<<<2048, 256, 0, stream>>>(bitmap, (size_t)NWORDS);
    zero_u32<<<2048, 256, 0, stream>>>((unsigned*)agg, (size_t)U * D);
    set_bits<<<(E + 255) / 256, 256, 0, stream>>>(row, col, cp, bitmap, E);
    scan_partial<<<NBLK, SCAN_TPB, 0, stream>>>(bitmap, bsums, NWORDS);
    scan_bsums<<<1, 32, 0, stream>>>(bsums, NBLK);
    scan_words<<<NBLK, SCAN_TPB, 0, stream>>>(bitmap, bsums, prefix, NWORDS);
    emit_edges<<<(NWORDS + 255) / 256, 256, 0, stream>>>(bitmap, prefix, cp, out, U, NWORDS);
    scatter_add<<<(E + 7) / 8, 256, 0, stream>>>(x, row, col, cp, bitmap, prefix, agg, E);

    int lb = (U + 127) / 128;
    if (lb > 0) {
        layer1_kernel<<<lb, 256, 0, stream>>>(agg, emb, W1, b1, hbuf, U);
        layer2_kernel<<<lb, 256, 0, stream>>>(hbuf, W2, b2, agg, U);
    }
}
